// XAIguiTransformerEncoderLayer_87806311399718
// MI455X (gfx1250) — matrix operand order repaired and checked by reading, not compiled, not run on hardware
//
#include <hip/hip_runtime.h>
#include <hip/hip_bf16.h>

// MI455X / gfx1250: wave32, WMMA 16x16x32 bf16 (fp32 accumulate).
typedef __attribute__((ext_vector_type(16))) __bf16 v16bf;
typedef __attribute__((ext_vector_type(8)))  float  v8f;

#define NSEQ  1024
#define CDIM  1024
#define FDIM  4096
#define HNUM  16
#define DHEAD 64
#define BDIM  8
#define MROWS (BDIM * NSEQ)

__device__ __forceinline__ v8f wmma_bf16(v16bf a, v16bf b, v8f c) {
  // 8 args: (neg_a, A, neg_b, B, c_mod, C, reuse_a, reuse_b)
  return __builtin_amdgcn_wmma_f32_16x16x32_bf16(false, a, false, b,
                                                 (short)0, c, false, false);
}

__device__ __forceinline__ float gelu_exact(float x) {
  return 0.5f * x * (1.0f + erff(x * 0.70710678118654752f));
}

// ---------------------------------------------------------------------------
// Generic GEMM: Out[M,N] = act(A[M,K] @ W[K,N] + bias)
// One wave per 32x64 output tile: two 16-row A fragments share every B
// fragment (weights), halving amortized weight traffic per WMMA.
//   A-frag layout: lane(m=l15) holds k = 16*half + e
//   B-frag layout: lane(n=l15) holds k = 16*half + e
//   D layout:      vgpr r -> row r + 8*half, col l15
// ---------------------------------------------------------------------------
template<int ACT>
__global__ __launch_bounds__(32)
void gemm_wmma(const float* __restrict__ A, const float* __restrict__ W,
               const float* __restrict__ bias, float* __restrict__ Out,
               int K, int ldw, int ldo) {
  const int lane = threadIdx.x;
  const int l15  = lane & 15;
  const int aoff = (lane >> 4) * 8;    // D row offset; also A prefetch offset
  const int boff = (lane >> 4) * 16;   // B-frag k offset
  const int mtile = blockIdx.x;        // 32 rows per tile
  const int ntile = blockIdx.y;        // 64 cols per tile
  const int colbase = ntile * 64;

  v8f acc[2][4];
  #pragma unroll
  for (int t = 0; t < 2; ++t)
    #pragma unroll
    for (int c = 0; c < 4; ++c)
      #pragma unroll
      for (int r = 0; r < 8; ++r) acc[t][c][r] = 0.0f;

  const float* arow0 = A + (size_t)(mtile * 32 + l15) * K;
  const float* arow1 = A + (size_t)(mtile * 32 + 16 + l15) * K;

  for (int kb = 0; kb < K; kb += 32) {
    if (kb + 32 < K) {                 // gfx1250 global_prefetch_b8 path
      __builtin_prefetch(arow0 + kb + 32 + aoff, 0, 1);
      __builtin_prefetch(arow1 + kb + 32 + aoff, 0, 1);
    }
    v16bf af0, af1;
    #pragma unroll
    for (int e = 0; e < 16; ++e) {
      const int k = kb + boff + e;
      af0[e] = (__bf16)arow0[k];
      af1[e] = (__bf16)arow1[k];
    }
    #pragma unroll
    for (int c = 0; c < 4; ++c) {
      const int col = colbase + c * 16 + l15;
      v16bf bfb;
      #pragma unroll
      for (int e = 0; e < 16; ++e)
        bfb[e] = (__bf16)W[(size_t)(kb + boff + e) * ldw + col];
      acc[0][c] = wmma_bf16(af0, bfb, acc[0][c]);
      acc[1][c] = wmma_bf16(af1, bfb, acc[1][c]);
    }
  }

  #pragma unroll
  for (int t = 0; t < 2; ++t)
    #pragma unroll
    for (int c = 0; c < 4; ++c) {
      const int col = colbase + c * 16 + l15;
      const float bv = bias ? bias[col] : 0.0f;
      #pragma unroll
      for (int r = 0; r < 8; ++r) {
        const int orow = mtile * 32 + t * 16 + r + aoff;
        float v = acc[t][c][r] + bv;
        if (ACT == 1) v = gelu_exact(v);
        Out[(size_t)orow * ldo + col] = v;
      }
    }
}

// ---------------------------------------------------------------------------
// drofe: rotary embedding with per-batch demographic scaling, in place on the
// packed (B,N,2C) q|k buffer. One thread per (b,n,h,pair) handling q and k.
// ---------------------------------------------------------------------------
__global__ __launch_bounds__(256)
void drofe_kernel(float* __restrict__ qk, const float* __restrict__ fb,
                  const float* __restrict__ demo) {
  const size_t idx = (size_t)blockIdx.x * 256 + threadIdx.x; // B*N*H*32 = 4M
  const int p = (int)(idx & 31);
  const int h = (int)((idx >> 5) & 15);
  const int n = (int)((idx >> 9) & 1023);
  const int b = (int)(idx >> 19);

  int j; float frb;
  if (p < 16) { j = p;      frb = fb[n * 2 + 0]; }
  else        { j = p - 16; frb = fb[n * 2 + 1]; }
  // freqs = linspace(1, MAX_FREQ/2=5, 16) * pi
  const float freq = (1.0f + 4.0f * (float)j / 15.0f) * 3.14159265358979323846f;
  const float ang = frb * freq;
  const float cs = cosf(ang), sn = sinf(ang);
  const float age = demo[b * 2 + 0], gen = demo[b * 2 + 1];
  const float ca = cs * age, sg = sn * gen;

  const size_t rowoff = ((size_t)b * NSEQ + n) * (2 * CDIM);
  {
    const size_t qi = rowoff + h * DHEAD + 2 * p;
    const float q0 = qk[qi], q1 = qk[qi + 1];
    qk[qi]     = q0 * ca - q1 * sg;   // q*cos*age + rotate_half(q)*sin*gender
    qk[qi + 1] = q1 * ca + q0 * sg;
  }
  {
    const size_t ki = rowoff + CDIM + h * DHEAD + 2 * p;
    const float k0 = qk[ki], k1 = qk[ki + 1];
    qk[ki]     = k0 * ca - k1 * sg;
    qk[ki + 1] = k1 * ca + k0 * sg;
  }
}

// ---------------------------------------------------------------------------
// Flash attention: one wave per (b, h, 16 query rows). Online softmax; scores
// never hit HBM. P (D-layout) -> LDS -> A-layout bf16 for the P@V WMMA.
// ---------------------------------------------------------------------------
__global__ __launch_bounds__(32)
void attn_wmma(const float* __restrict__ qkbuf, const float* __restrict__ vmat,
               float* __restrict__ o) {
  const int lane = threadIdx.x;
  const int l15  = lane & 15;
  const int aoff = (lane >> 4) * 8;
  const int boff = (lane >> 4) * 16;
  const int qt = blockIdx.x, h = blockIdx.y, b = blockIdx.z;

  const float* qbase = qkbuf + (size_t)b * NSEQ * (2 * CDIM) + h * DHEAD;
  const float* kbase = qbase + CDIM;
  const float* vbase = vmat  + (size_t)b * NSEQ * CDIM + h * DHEAD;

  const int qrow = qt * 16 + l15;
  v16bf qa0, qa1;                               // Q rows, depth 0..31 / 32..63
  #pragma unroll
  for (int e = 0; e < 16; ++e) {
    qa0[e] = (__bf16)qbase[(size_t)qrow * (2 * CDIM) + boff + e];
    qa1[e] = (__bf16)qbase[(size_t)qrow * (2 * CDIM) + 32 + boff + e];
  }

  float mrow[8], lrow[8];
  v8f oacc[4];
  #pragma unroll
  for (int r = 0; r < 8; ++r) { mrow[r] = -1e30f; lrow[r] = 0.0f; }
  #pragma unroll
  for (int c = 0; c < 4; ++c)
    #pragma unroll
    for (int r = 0; r < 8; ++r) oacc[c][r] = 0.0f;

  __shared__ __bf16 psh[16 * 32];
  const float scale = 0.125f;                   // D^-0.5 = 64^-0.5

  for (int jb = 0; jb < NSEQ; jb += 32) {
    v8f s0, s1;
    #pragma unroll
    for (int r = 0; r < 8; ++r) { s0[r] = 0.0f; s1[r] = 0.0f; }
    {
      const int col = jb + l15;                 // K^T columns = K rows
      v16bf kb0, kb1;
      #pragma unroll
      for (int e = 0; e < 16; ++e) {
        kb0[e] = (__bf16)kbase[(size_t)col * (2 * CDIM) + boff + e];
        kb1[e] = (__bf16)kbase[(size_t)col * (2 * CDIM) + 32 + boff + e];
      }
      s0 = wmma_bf16(qa0, kb0, s0);
      s0 = wmma_bf16(qa1, kb1, s0);
    }
    {
      const int col = jb + 16 + l15;
      v16bf kb0, kb1;
      #pragma unroll
      for (int e = 0; e < 16; ++e) {
        kb0[e] = (__bf16)kbase[(size_t)col * (2 * CDIM) + boff + e];
        kb1[e] = (__bf16)kbase[(size_t)col * (2 * CDIM) + 32 + boff + e];
      }
      s1 = wmma_bf16(qa0, kb0, s1);
      s1 = wmma_bf16(qa1, kb1, s1);
    }

    // Online softmax over this 16x32 score block (row = r + aoff, spread
    // across one 16-lane half -> butterfly reduce with masks 1,2,4,8).
    float fr[8];
    #pragma unroll
    for (int r = 0; r < 8; ++r) {
      const float a0 = s0[r] * scale, a1 = s1[r] * scale;
      float tm = fmaxf(a0, a1);
      tm = fmaxf(tm, __shfl_xor(tm, 1, 32));
      tm = fmaxf(tm, __shfl_xor(tm, 2, 32));
      tm = fmaxf(tm, __shfl_xor(tm, 4, 32));
      tm = fmaxf(tm, __shfl_xor(tm, 8, 32));
      const float mn = fmaxf(mrow[r], tm);
      const float p0 = expf(a0 - mn), p1 = expf(a1 - mn);
      float ts = p0 + p1;
      ts += __shfl_xor(ts, 1, 32);
      ts += __shfl_xor(ts, 2, 32);
      ts += __shfl_xor(ts, 4, 32);
      ts += __shfl_xor(ts, 8, 32);
      const float f = expf(mrow[r] - mn);
      lrow[r] = lrow[r] * f + ts;
      mrow[r] = mn;
      fr[r] = f;
      psh[(r + aoff) * 32 + l15]      = (__bf16)p0;   // D-layout -> [m][k]
      psh[(r + aoff) * 32 + 16 + l15] = (__bf16)p1;
    }
    #pragma unroll
    for (int c = 0; c < 4; ++c)
      #pragma unroll
      for (int r = 0; r < 8; ++r) oacc[c][r] *= fr[r];

    __syncthreads();
    v16bf pa;                                   // P as A-frag (16x32)
    #pragma unroll
    for (int e = 0; e < 16; ++e)
      pa[e] = psh[l15 * 32 + boff + e];
    __syncthreads();

    #pragma unroll
    for (int c = 0; c < 4; ++c) {               // O += P @ V, 4 d-chunks of 16
      v16bf vb;
      #pragma unroll
      for (int e = 0; e < 16; ++e) {
        const int kr = jb + boff + e;
        vb[e] = (__bf16)vbase[(size_t)kr * CDIM + c * 16 + l15];
      }
      oacc[c] = wmma_bf16(pa, vb, oacc[c]);
    }
  }

  #pragma unroll
  for (int c = 0; c < 4; ++c)
    #pragma unroll
    for (int r = 0; r < 8; ++r) {
      const int orow = qt * 16 + r + aoff;
      o[((size_t)b * NSEQ + orow) * CDIM + h * DHEAD + c * 16 + l15] =
          oacc[c][r] / lrow[r];
    }
}

// ---------------------------------------------------------------------------
// Fused residual + LayerNorm: out = LN(base + gamma*delta; w, b).
// One 256-thread block per row of C=1024. base/out may alias (row-local).
// ---------------------------------------------------------------------------
__global__ __launch_bounds__(256)
void resid_ln(const float* base, const float* __restrict__ delta,
              const float* __restrict__ gamma, const float* __restrict__ w,
              const float* __restrict__ bvec, float* out) {
  const int row = blockIdx.x;
  const int t = threadIdx.x;
  __shared__ float red[256];
  __shared__ float red2[256];
  float vals[4];
  float s = 0.0f, s2 = 0.0f;
  #pragma unroll
  for (int i = 0; i < 4; ++i) {
    const int c = t + i * 256;
    const float v = base[(size_t)row * CDIM + c] +
                    gamma[c] * delta[(size_t)row * CDIM + c];
    vals[i] = v; s += v; s2 += v * v;
  }
  red[t] = s; red2[t] = s2;
  __syncthreads();
  for (int off = 128; off > 0; off >>= 1) {
    if (t < off) { red[t] += red[t + off]; red2[t] += red2[t + off]; }
    __syncthreads();
  }
  const float mu  = red[0] * (1.0f / CDIM);
  const float var = red2[0] * (1.0f / CDIM) - mu * mu;
  const float inv = rsqrtf(var + 1e-5f);
  #pragma unroll
  for (int i = 0; i < 4; ++i) {
    const int c = t + i * 256;
    out[(size_t)row * CDIM + c] = (vals[i] - mu) * inv * w[c] + bvec[c];
  }
}

// ---------------------------------------------------------------------------
extern "C" void kernel_launch(void* const* d_in, const int* in_sizes, int n_in,
                              void* d_out, int out_size, void* d_ws, size_t ws_size,
                              hipStream_t stream) {
  (void)in_sizes; (void)n_in; (void)out_size; (void)ws_size;
  const float* x      = (const float*)d_in[0];
  const float* demo   = (const float*)d_in[1];
  const float* expl   = (const float*)d_in[2];
  const float* fb     = (const float*)d_in[3];
  const float* w_qkv  = (const float*)d_in[4];
  const float* w_proj = (const float*)d_in[5];
  const float* b_proj = (const float*)d_in[6];
  const float* gamma1 = (const float*)d_in[7];
  const float* gamma2 = (const float*)d_in[8];
  const float* ln1w   = (const float*)d_in[9];
  const float* ln1b   = (const float*)d_in[10];
  const float* ln2w   = (const float*)d_in[11];
  const float* ln2b   = (const float*)d_in[12];
  const float* w1     = (const float*)d_in[13];
  const float* b1     = (const float*)d_in[14];
  const float* w2     = (const float*)d_in[15];
  const float* b2     = (const float*)d_in[16];
  float* out = (float*)d_out;
  float* ws  = (float*)d_ws;

  const size_t MF = 1048576;         // 1M floats
  float* vbuf  = ws;                 // [0, 8M)   : V (B,N,C)
  float* qkbuf = ws + 8 * MF;        // [8M, 24M) : Q|K (B,N,2C)
  float* obuf  = ws + 24 * MF;       // [24M,32M) : attention out
  float* pbuf  = ws + 32 * MF;       // [32M,40M) : proj out
  float* hbuf  = ws + 8 * MF;        // [8M, 40M) : MLP hidden (reuses dead bufs)
  float* h2buf = ws;                 // [0, 8M)   : MLP out (reuses vbuf)

  // 1) V = x @ w_qkv[:, 2C:3C]
  gemm_wmma<0><<<dim3(MROWS / 32, CDIM / 64), 32, 0, stream>>>(
      x, w_qkv + 2 * CDIM, nullptr, vbuf, CDIM, 3 * CDIM, CDIM);
  // 2) Q|K = explanation @ w_qkv[:, 0:2C]
  gemm_wmma<0><<<dim3(MROWS / 32, (2 * CDIM) / 64), 32, 0, stream>>>(
      expl, w_qkv, nullptr, qkbuf, CDIM, 3 * CDIM, 2 * CDIM);
  // 3) drofe rotary (in place)
  drofe_kernel<<<(BDIM * NSEQ * HNUM * 32) / 256, 256, 0, stream>>>(qkbuf, fb, demo);
  // 4) flash attention -> obuf
  attn_wmma<<<dim3(NSEQ / 16, HNUM, BDIM), 32, 0, stream>>>(qkbuf, vbuf, obuf);
  // 5) out projection
  gemm_wmma<0><<<dim3(MROWS / 32, CDIM / 64), 32, 0, stream>>>(
      obuf, w_proj, b_proj, pbuf, CDIM, CDIM, CDIM);
  // 6) x = LN(x + gamma1 * proj) -> d_out
  resid_ln<<<MROWS, 256, 0, stream>>>(x, pbuf, gamma1, ln1w, ln1b, out);
  // 7) h = gelu(x @ w1 + b1)
  gemm_wmma<1><<<dim3(MROWS / 32, FDIM / 64), 32, 0, stream>>>(
      out, w1, b1, hbuf, CDIM, FDIM, FDIM);
  // 8) h2 = h @ w2 + b2
  gemm_wmma<0><<<dim3(MROWS / 32, CDIM / 64), 32, 0, stream>>>(
      hbuf, w2, b2, h2buf, FDIM, CDIM, CDIM);
  // 9) out = LN(out + gamma2 * h2)
  resid_ln<<<MROWS, 256, 0, stream>>>(out, h2buf, gamma2, ln2w, ln2b, out);
}
